// PerformerSelfAttention_42082089566413
// MI455X (gfx1250) — matrix operand order repaired and checked by reading, not compiled, not run on hardware
//
#include <hip/hip_runtime.h>
#include <hip/hip_bf16.h>
#include <math.h>

// ---------------------------------------------------------------------------
// Performer self-attention + FF block for MI455X (gfx1250), wave32 + WMMA bf16.
// GEMMs: C = A(row-major MxK) @ W(N,K row-major)^T via v_wmma_f32_16x16x32_bf16.
// Block tiles are staged into LDS with CDNA5 async copies (ASYNCcnt) when the
// toolchain exposes them, double-buffered so copies overlap WMMA.
// ---------------------------------------------------------------------------

typedef unsigned short ushort_t;
typedef __attribute__((ext_vector_type(16))) __bf16          v16bf;
typedef __attribute__((ext_vector_type(16))) unsigned short  v16us;
typedef __attribute__((ext_vector_type(8)))  unsigned short  v8us;
typedef __attribute__((ext_vector_type(8)))  float           v8f;
typedef __attribute__((ext_vector_type(4)))  int             v4i;

union Frag {
  v16bf bf;
  v16us us;
  struct { v8us lo, hi; } h;
};

#define BB   8
#define NNs  4096
#define DDm  512
#define HHn  4
#define DHd  128
#define MMf  256
#define FFd  2048
#define BNr  32768   // BB*NNs
#define ZZb  32      // BB*HHn

#define LDS_STRIDE 40   // 32 bf16 + 8 pad => 80B rows, conflict-free b128 reads

#if defined(__has_builtin)
# if __has_builtin(__builtin_amdgcn_global_load_async_to_lds_b128) && \
     __has_builtin(__builtin_amdgcn_s_wait_asynccnt)
#  define PERF_ASYNC_LDS 1
# endif
#endif
#ifndef PERF_ASYNC_LDS
# define PERF_ASYNC_LDS 0
#endif

#define AS1 __attribute__((address_space(1)))
#define AS3 __attribute__((address_space(3)))

__device__ __forceinline__ ushort_t f2bf(float f) {
  unsigned int u = __float_as_uint(f);
  u += 0x7fffu + ((u >> 16) & 1u);        // round-to-nearest-even
  return (ushort_t)(u >> 16);
}
__device__ __forceinline__ float bf2f(ushort_t s) {
  return __uint_as_float(((unsigned int)s) << 16);
}

// ---------------------------------------------------------------------------
// Generic WMMA GEMM.  256 threads = 8 waves; wave tile 32x32 (2x2 WMMA tiles),
// block tile 128(M) x 64(N).  grid = (N/64, M/128, Z), Z decomposed (b,h).
// Per K=32 step the block stages A(128x32) + B(64x32) bf16 into LDS once
// (async, double-buffered) and waves read fragments with ds_load_b128:
//   A frag: lane L<16 -> row L, k {0..7,16..23}; L>=16 -> row L-16, k {8..15,24..31}
//   B frag: lane L<16 -> col L, k {0..7,16..23}; L>=16 -> col L-16, k {8..15,24..31}
//   C/D f32: VGPR r -> (m = 8*(L>>4)+r, n = L&15)
// ---------------------------------------------------------------------------
enum {
  EPI_BF16 = 0,         // store bf16
  EPI_VT,               // store bf16 transposed per head (vT[b,h,d,n])
  EPI_F32,              // store fp32
  EPI_BIAS_RES_F32,     // fp32: acc + bias[col] + res[row,col]
  EPI_GELU_BF16,        // bf16: gelu(acc + bias[col])  (exact erf gelu)
  EPI_BIAS_RES_BF16,    // bf16: acc + bias[col] + res[row,col]
  EPI_BIAS_F32,         // fp32: acc + bias[col]
  EPI_SCALEROW_BF16     // bf16: acc * P1[row]   (d_inv scaling)
};

template <int EPI>
__global__ __launch_bounds__(256) void gemm_k(
    const ushort_t* __restrict__ A, long sAb, long sAh, int lda,
    const ushort_t* __restrict__ W, long sWb, long sWh,
    void* __restrict__ Cp, long sCb, long sCh, int ldc,
    const float* __restrict__ P1, long sP1b, long sP1h,
    const float* __restrict__ P2,
    int K, int HZ)
{
  __shared__ __align__(64) ushort_t smA[2][128 * LDS_STRIDE];
  __shared__ __align__(64) ushort_t smB[2][64 * LDS_STRIDE];

  const int tid  = threadIdx.x;
  const int lane = tid & 31;
  const int wid  = tid >> 5;
  const int la   = lane & 15;
  const int lh   = lane >> 4;
  const int mW   = (wid >> 1) * 32;          // wave M offset inside block tile
  const int nW   = (wid & 1) * 32;           // wave N offset inside block tile
  const int m0   = blockIdx.y * 128 + mW;
  const int n0   = blockIdx.x * 64 + nW;
  const int z    = blockIdx.z;
  const int zb   = z / HZ;
  const int zh   = z % HZ;

  const ushort_t* Ab = A + (long)zb * sAb + (long)zh * sAh;
  const ushort_t* Wb = W + (long)zb * sWb + (long)zh * sWh;

  // --- staging assignment: A tile = 512 x 16B chunks (2/thread), B = 1/thread
  const int ca0 = tid, ca1 = tid + 256;
  const int ra0 = ca0 >> 2, sa0 = (ca0 & 3) * 8;
  const int ra1 = ca1 >> 2, sa1 = (ca1 & 3) * 8;
  const int rb  = tid >> 2, sb  = (tid & 3) * 8;
  const ushort_t* gA0 = Ab + (long)(blockIdx.y * 128 + ra0) * lda + sa0;
  const ushort_t* gA1 = Ab + (long)(blockIdx.y * 128 + ra1) * lda + sa1;
  const ushort_t* gB  = Wb + (long)(blockIdx.x * 64 + rb) * K + sb;
  const int lA0 = ra0 * LDS_STRIDE + sa0;
  const int lA1 = ra1 * LDS_STRIDE + sa1;
  const int lB  = rb  * LDS_STRIDE + sb;

  auto stage = [&](int sbuf, int k0) {
#if PERF_ASYNC_LDS
    __builtin_amdgcn_global_load_async_to_lds_b128(
        (AS1 v4i*)(gA0 + k0), (AS3 v4i*)(&smA[sbuf][lA0]), 0, 0);
    __builtin_amdgcn_global_load_async_to_lds_b128(
        (AS1 v4i*)(gA1 + k0), (AS3 v4i*)(&smA[sbuf][lA1]), 0, 0);
    __builtin_amdgcn_global_load_async_to_lds_b128(
        (AS1 v4i*)(gB + k0), (AS3 v4i*)(&smB[sbuf][lB]), 0, 0);
#else
    v8us t0 = *(const v8us*)(gA0 + k0);
    v8us t1 = *(const v8us*)(gA1 + k0);
    v8us t2 = *(const v8us*)(gB + k0);
    *(v8us*)(&smA[sbuf][lA0]) = t0;
    *(v8us*)(&smA[sbuf][lA1]) = t1;
    *(v8us*)(&smB[sbuf][lB]) = t2;
#endif
  };

  v8f acc[2][2] = {};
  int buf = 0;
  stage(0, 0);

  for (int k0 = 0; k0 < K; k0 += 32) {
    const bool hasNext = (k0 + 32) < K;
    if (hasNext) {
      stage(buf ^ 1, k0 + 32);
      if (k0 + 64 < K) {                     // gfx1250 global_prefetch_b8
        __builtin_prefetch(gA0 + k0 + 64, 0, 1);
        __builtin_prefetch(gB + k0 + 64, 0, 1);
      }
    }
#if PERF_ASYNC_LDS
    if (hasNext) __builtin_amdgcn_s_wait_asynccnt(3);
    else         __builtin_amdgcn_s_wait_asynccnt(0);
#endif
    __syncthreads();

    Frag a0, a1, b0, b1;
    const ushort_t* pa = &smA[buf][(mW + la) * LDS_STRIDE + lh * 8];
    const ushort_t* pb = &smB[buf][(nW + la) * LDS_STRIDE + lh * 8];
    a0.h.lo = *(const v8us*)(pa);
    a0.h.hi = *(const v8us*)(pa + 16);
    a1.h.lo = *(const v8us*)(pa + 16 * LDS_STRIDE);
    a1.h.hi = *(const v8us*)(pa + 16 * LDS_STRIDE + 16);
    b0.h.lo = *(const v8us*)(pb);
    b0.h.hi = *(const v8us*)(pb + 16);
    b1.h.lo = *(const v8us*)(pb + 16 * LDS_STRIDE);
    b1.h.hi = *(const v8us*)(pb + 16 * LDS_STRIDE + 16);

    acc[0][0] = __builtin_amdgcn_wmma_f32_16x16x32_bf16(
        false, a0.bf, false, b0.bf, (short)0, acc[0][0], false, false);
    acc[0][1] = __builtin_amdgcn_wmma_f32_16x16x32_bf16(
        false, a0.bf, false, b1.bf, (short)0, acc[0][1], false, false);
    acc[1][0] = __builtin_amdgcn_wmma_f32_16x16x32_bf16(
        false, a1.bf, false, b0.bf, (short)0, acc[1][0], false, false);
    acc[1][1] = __builtin_amdgcn_wmma_f32_16x16x32_bf16(
        false, a1.bf, false, b1.bf, (short)0, acc[1][1], false, false);

    __syncthreads();
    buf ^= 1;
  }

#pragma unroll
  for (int i = 0; i < 2; ++i) {
#pragma unroll
    for (int j = 0; j < 2; ++j) {
      const int col  = n0 + j * 16 + la;
      const int rowb = m0 + i * 16 + lh * 8;
#pragma unroll
      for (int r = 0; r < 8; ++r) {
        const int row = rowb + r;
        float v = acc[i][j][r];
        if (EPI == EPI_BF16) {
          ushort_t* C = (ushort_t*)Cp + (long)zb * sCb + (long)zh * sCh;
          C[(long)row * ldc + col] = f2bf(v);
        } else if (EPI == EPI_VT) {
          // v tile: row = b*4096+n, col = h*128+d  ->  vT[b,h,d,n]
          ushort_t* C = (ushort_t*)Cp;
          long bb = row >> 12, n = row & 4095;
          int hh = col >> 7, d = col & 127;
          C[((bb * HHn + hh) * (long)DHd + d) * NNs + n] = f2bf(v);
        } else if (EPI == EPI_F32) {
          float* C = (float*)Cp + (long)zb * sCb + (long)zh * sCh;
          C[(long)row * ldc + col] = v;
        } else if (EPI == EPI_BIAS_RES_F32) {
          float* C = (float*)Cp;
          C[(long)row * ldc + col] = v + P1[col] + P2[(long)row * ldc + col];
        } else if (EPI == EPI_GELU_BF16) {
          ushort_t* C = (ushort_t*)Cp;
          float t = v + P1[col];
          C[(long)row * ldc + col] =
              f2bf(0.5f * t * (1.0f + erff(t * 0.70710678118654752f)));
        } else if (EPI == EPI_BIAS_RES_BF16) {
          ushort_t* C = (ushort_t*)Cp;
          C[(long)row * ldc + col] =
              f2bf(v + P1[col] + P2[(long)row * ldc + col]);
        } else if (EPI == EPI_BIAS_F32) {
          float* C = (float*)Cp;
          C[(long)row * ldc + col] = v + P1[col];
        } else if (EPI == EPI_SCALEROW_BF16) {
          ushort_t* C = (ushort_t*)Cp + (long)zb * sCb + (long)zh * sCh;
          float s = P1[(long)zb * sP1b + (long)zh * sP1h + row];
          C[(long)row * ldc + col] = f2bf(v * s);
        }
      }
    }
  }
}

// ------------------------- elementwise / reduction kernels ------------------

__global__ __launch_bounds__(256) void cvt_k(const float* __restrict__ s,
                                             ushort_t* __restrict__ d,
                                             long n, float sc) {
  long i = (long)blockIdx.x * 256 + threadIdx.x;
  if (i < n) d[i] = f2bf(s[i] * sc);
}

// LayerNorm over 512 cols, one wave per row (8 rows / block).
__global__ __launch_bounds__(256) void ln_k(const float* __restrict__ x,
                                            const float* __restrict__ g,
                                            const float* __restrict__ be,
                                            ushort_t* __restrict__ o) {
  int lane = threadIdx.x & 31, wid = threadIdx.x >> 5;
  long row = (long)blockIdx.x * 8 + wid;
  const float* xr = x + row * DDm;
  float v[16], s = 0.f;
#pragma unroll
  for (int t = 0; t < 16; ++t) { v[t] = xr[lane + 32 * t]; s += v[t]; }
  for (int off = 16; off; off >>= 1) s += __shfl_xor(s, off, 32);
  float mu = s * (1.f / 512.f), s2 = 0.f;
#pragma unroll
  for (int t = 0; t < 16; ++t) { float d = v[t] - mu; s2 += d * d; }
  for (int off = 16; off; off >>= 1) s2 += __shfl_xor(s2, off, 32);
  float r = rsqrtf(s2 * (1.f / 512.f) + 1e-5f);
#pragma unroll
  for (int t = 0; t < 16; ++t) {
    int c = lane + 32 * t;
    o[row * DDm + c] = f2bf((v[t] - mu) * r * g[c] + be[c]);
  }
}

// diag = 0.5*dn^2 * sum(q^2) per (row, head); one wave per pair, q & k fused.
__global__ __launch_bounds__(256) void diag_k(const ushort_t* __restrict__ q,
                                              const ushort_t* __restrict__ k,
                                              float* __restrict__ dq,
                                              float* __restrict__ dk) {
  int lane = threadIdx.x & 31, wid = threadIdx.x >> 5;
  long p = (long)blockIdx.x * 8 + wid;   // [0, BN*H)
  long row = p >> 2;
  int h = (int)(p & 3);
  long base = row * DDm + (long)h * DHd;
  float sq = 0.f, sk = 0.f;
#pragma unroll
  for (int t = 0; t < 4; ++t) {
    float a = bf2f(q[base + lane + 32 * t]);
    float b = bf2f(k[base + lane + 32 * t]);
    sq += a * a; sk += b * b;
  }
  for (int off = 16; off; off >>= 1) {
    sq += __shfl_xor(sq, off, 32);
    sk += __shfl_xor(sk, off, 32);
  }
  if (lane == 0) {
    long o = ((row >> 12) * HHn + h) * (long)NNs + (row & 4095);
    dq[o] = sq * 0.044194173824159216f;   // 0.5 / sqrt(128)
    dk[o] = sk * 0.044194173824159216f;
  }
}

// qp = ratio*(exp(qdash - diag - rowmax) + eps), per-row stabilizer.
__global__ __launch_bounds__(256) void qp_k(const float* __restrict__ qd,
                                            const float* __restrict__ diag,
                                            ushort_t* __restrict__ qp) {
  int lane = threadIdx.x & 31, wid = threadIdx.x >> 5;
  long p = (long)blockIdx.x * 8 + wid;   // z*N + n
  const float* row = qd + p * MMf;
  float v[8], mx = -INFINITY;
#pragma unroll
  for (int t = 0; t < 8; ++t) { v[t] = row[lane + 32 * t]; mx = fmaxf(mx, v[t]); }
  for (int off = 16; off; off >>= 1) mx = fmaxf(mx, __shfl_xor(mx, off, 32));
  float d = diag[p];
#pragma unroll
  for (int t = 0; t < 8; ++t)
    qp[p * MMf + lane + 32 * t] = f2bf(0.0625f * (expf(v[t] - d - mx) + 1e-4f));
}

__global__ __launch_bounds__(256) void rowmax_k(const float* __restrict__ kd,
                                                float* __restrict__ rm) {
  int lane = threadIdx.x & 31, wid = threadIdx.x >> 5;
  long p = (long)blockIdx.x * 8 + wid;
  const float* row = kd + p * MMf;
  float mx = -INFINITY;
#pragma unroll
  for (int t = 0; t < 8; ++t) mx = fmaxf(mx, row[lane + 32 * t]);
  for (int off = 16; off; off >>= 1) mx = fmaxf(mx, __shfl_xor(mx, off, 32));
  if (lane == 0) rm[p] = mx;
}

__global__ __launch_bounds__(256) void stabk_k(const float* __restrict__ rm,
                                               float* __restrict__ stab) {
  __shared__ float sm[256];
  int z = blockIdx.x;
  float m = -INFINITY;
  for (int i = threadIdx.x; i < NNs; i += 256) m = fmaxf(m, rm[(long)z * NNs + i]);
  sm[threadIdx.x] = m;
  __syncthreads();
  for (int s = 128; s > 0; s >>= 1) {
    if ((int)threadIdx.x < s)
      sm[threadIdx.x] = fmaxf(sm[threadIdx.x], sm[threadIdx.x + s]);
    __syncthreads();
  }
  if (threadIdx.x == 0) stab[z] = sm[0];
}

// kpT[z,m,n] = ratio*(exp(kdash[z,n,m] - diag - stab[z]) + eps)  (transposed)
__global__ __launch_bounds__(256) void kpt_k(const float* __restrict__ kd,
                                             const float* __restrict__ diag,
                                             const float* __restrict__ stab,
                                             ushort_t* __restrict__ kpT) {
  long idx = (long)blockIdx.x * 256 + threadIdx.x;   // (z,n,m) packed
  int m = (int)(idx & 255);
  long nz = idx >> 8;                                // z*N + n
  int z = (int)(nz >> 12);
  long n = nz & 4095;
  float v = 0.0625f * (expf(kd[idx] - diag[nz] - stab[z]) + 1e-4f);
  kpT[((long)z * MMf + m) * NNs + n] = f2bf(v);
}

// ksum[z,m] = sum_n kpT[z,m,n]   (row sums of kpT, coalesced)
__global__ __launch_bounds__(256) void ksum_k(const ushort_t* __restrict__ kpT,
                                              float* __restrict__ ks) {
  int lane = threadIdx.x & 31, wid = threadIdx.x >> 5;
  long p = (long)blockIdx.x * 8 + wid;   // z*M + m
  const ushort_t* row = kpT + p * NNs;
  float s = 0.f;
  for (int t = 0; t < 128; ++t) s += bf2f(row[lane + 32 * t]);
  for (int off = 16; off; off >>= 1) s += __shfl_xor(s, off, 32);
  if (lane == 0) ks[p] = s;
}

// dinv[z,n] = 1 / dot(qp[z,n,:], ksum[z,:])
__global__ __launch_bounds__(256) void dinv_k(const ushort_t* __restrict__ qp,
                                              const float* __restrict__ ks,
                                              float* __restrict__ dinv) {
  int lane = threadIdx.x & 31, wid = threadIdx.x >> 5;
  long p = (long)blockIdx.x * 8 + wid;   // z*N + n
  long z = p >> 12;
  float s = 0.f;
#pragma unroll
  for (int t = 0; t < 8; ++t) {
    int e = lane + 32 * t;
    s += bf2f(qp[p * MMf + e]) * ks[z * MMf + e];
  }
  for (int off = 16; off; off >>= 1) s += __shfl_xor(s, off, 32);
  if (lane == 0) dinv[p] = 1.0f / s;
}

// ---------------------------------------------------------------------------

extern "C" void kernel_launch(void* const* d_in, const int* in_sizes, int n_in,
                              void* d_out, int out_size, void* d_ws, size_t ws_size,
                              hipStream_t stream) {
  const float* x    = (const float*)d_in[0];
  const float* proj = (const float*)d_in[1];
  const float* g1   = (const float*)d_in[2];
  const float* bln1 = (const float*)d_in[3];
  const float* wq   = (const float*)d_in[4];
  const float* wk   = (const float*)d_in[5];
  const float* wv   = (const float*)d_in[6];
  const float* wo   = (const float*)d_in[7];
  const float* bo   = (const float*)d_in[8];
  const float* g2   = (const float*)d_in[9];
  const float* bln2 = (const float*)d_in[10];
  const float* w1   = (const float*)d_in[11];
  const float* pb1  = (const float*)d_in[12];
  const float* w2   = (const float*)d_in[13];
  const float* pb2  = (const float*)d_in[14];
  const float* wop  = (const float*)d_in[15];
  const float* bop  = (const float*)d_in[16];
  float* out = (float*)d_out;

  char* ws = (char*)d_ws;
  size_t cur = 0;
  auto take = [&](size_t bytes) -> char* {
    size_t a = (cur + 255) & ~(size_t)255;
    cur = a + bytes;
    return ws + a;
  };

  ushort_t* wq_b  = (ushort_t*)take((size_t)DDm * DDm * 2);
  ushort_t* wk_b  = (ushort_t*)take((size_t)DDm * DDm * 2);
  ushort_t* wv_b  = (ushort_t*)take((size_t)DDm * DDm * 2);
  ushort_t* wo_b  = (ushort_t*)take((size_t)DDm * DDm * 2);
  ushort_t* wop_b = (ushort_t*)take((size_t)DDm * DDm * 2);
  ushort_t* w1_b  = (ushort_t*)take((size_t)FFd * DDm * 2);
  ushort_t* w2_b  = (ushort_t*)take((size_t)DDm * FFd * 2);
  ushort_t* pj_b  = (ushort_t*)take((size_t)MMf * DHd * 2);

  ushort_t* h_b   = (ushort_t*)take((size_t)BNr * DDm * 2);
  ushort_t* q_b   = (ushort_t*)take((size_t)BNr * DDm * 2);
  ushort_t* k_b   = (ushort_t*)take((size_t)BNr * DDm * 2);
  ushort_t* vT_b  = (ushort_t*)take((size_t)ZZb * DHd * NNs * 2);
  float*    dq    = (float*)take((size_t)ZZb * NNs * 4);
  float*    dk    = (float*)take((size_t)ZZb * NNs * 4);
  float*    qdash = (float*)take((size_t)ZZb * NNs * MMf * 4);
  float*    kdash = (float*)take((size_t)ZZb * NNs * MMf * 4);
  ushort_t* qp_b  = (ushort_t*)take((size_t)ZZb * NNs * MMf * 2);
  ushort_t* kpT_b = (ushort_t*)take((size_t)ZZb * MMf * NNs * 2);
  float*    rmax  = (float*)take((size_t)ZZb * NNs * 4);
  float*    stbk  = (float*)take(256);
  float*    ksum  = (float*)take((size_t)ZZb * MMf * 4);
  float*    dinv  = (float*)take((size_t)ZZb * NNs * 4);
  ushort_t* ctxT  = (ushort_t*)take((size_t)ZZb * DHd * MMf * 2);
  ushort_t* ff1_b = (ushort_t*)take((size_t)BNr * FFd * 2);

  // Aliases (lifetimes do not overlap):
  float*    x1    = kdash;               // fp32 [BN,512]  (kdash dead)
  ushort_t* h2_b  = (ushort_t*)qdash;    // bf16 [BN,512]  (qdash dead)
  ushort_t* attn_b = k_b;                // bf16 [BN,512]  (k_b dead)
  ushort_t* x2_b   = q_b;                // bf16 [BN,512]  (q_b dead)

  const float dn = 0.29730177875068026f; // 128^-0.25, folded into proj

  auto cvt = [&](const float* s, ushort_t* d, long n, float sc) {
    cvt_k<<<dim3((unsigned)((n + 255) / 256)), dim3(256), 0, stream>>>(s, d, n, sc);
  };
  cvt(wq, wq_b, (long)DDm * DDm, 1.f);
  cvt(wk, wk_b, (long)DDm * DDm, 1.f);
  cvt(wv, wv_b, (long)DDm * DDm, 1.f);
  cvt(wo, wo_b, (long)DDm * DDm, 1.f);
  cvt(wop, wop_b, (long)DDm * DDm, 1.f);
  cvt(w1, w1_b, (long)FFd * DDm, 1.f);
  cvt(w2, w2_b, (long)DDm * FFd, 1.f);
  cvt(proj, pj_b, (long)MMf * DHd, dn);

  // LN1
  ln_k<<<dim3(BNr / 8), dim3(256), 0, stream>>>(x, g1, bln1, h_b);

  // QKV GEMMs: [32768,512] x [512,512]^T
  dim3 gQKV(DDm / 64, BNr / 128, 1);
  gemm_k<EPI_BF16><<<gQKV, 256, 0, stream>>>(h_b, 0, 0, DDm, wq_b, 0, 0,
      q_b, 0, 0, DDm, nullptr, 0, 0, nullptr, DDm, 1);
  gemm_k<EPI_BF16><<<gQKV, 256, 0, stream>>>(h_b, 0, 0, DDm, wk_b, 0, 0,
      k_b, 0, 0, DDm, nullptr, 0, 0, nullptr, DDm, 1);
  gemm_k<EPI_VT><<<gQKV, 256, 0, stream>>>(h_b, 0, 0, DDm, wv_b, 0, 0,
      vT_b, 0, 0, 0, nullptr, 0, 0, nullptr, DDm, 1);

  // diag (sum of squares per row/head) for q and k
  diag_k<<<dim3((BNr * HHn) / 8), 256, 0, stream>>>(q_b, k_b, dq, dk);

  // Feature projections: per (b,h): [4096,128] x [256,128]^T -> fp32 [z,n,256]
  dim3 gFP(MMf / 64, NNs / 128, ZZb);
  gemm_k<EPI_F32><<<gFP, 256, 0, stream>>>(
      q_b, (long)NNs * DDm, DHd, DDm, pj_b, 0, 0,
      qdash, (long)HHn * NNs * MMf, (long)NNs * MMf, MMf,
      nullptr, 0, 0, nullptr, DHd, HHn);
  gemm_k<EPI_F32><<<gFP, 256, 0, stream>>>(
      k_b, (long)NNs * DDm, DHd, DDm, pj_b, 0, 0,
      kdash, (long)HHn * NNs * MMf, (long)NNs * MMf, MMf,
      nullptr, 0, 0, nullptr, DHd, HHn);

  // qp (row-stabilized exp) ; kp stabilizer (global per z) + transposed kp
  qp_k<<<dim3((ZZb * NNs) / 8), 256, 0, stream>>>(qdash, dq, qp_b);
  rowmax_k<<<dim3((ZZb * NNs) / 8), 256, 0, stream>>>(kdash, rmax);
  stabk_k<<<dim3(ZZb), 256, 0, stream>>>(rmax, stbk);
  kpt_k<<<dim3((unsigned)(((long)ZZb * NNs * MMf) / 256)), 256, 0, stream>>>(
      kdash, dk, stbk, kpT_b);
  ksum_k<<<dim3((ZZb * MMf) / 8), 256, 0, stream>>>(kpT_b, ksum);
  dinv_k<<<dim3((ZZb * NNs) / 8), 256, 0, stream>>>(qp_b, ksum, dinv);

  // ctxT[z,d,m] = sum_n vT[z,d,n] * kpT[z,m,n] : [128,4096] x [256,4096]^T
  dim3 gCTX(MMf / 64, 1, ZZb);
  gemm_k<EPI_BF16><<<gCTX, 256, 0, stream>>>(
      vT_b, (long)HHn * DHd * NNs, (long)DHd * NNs, NNs,
      kpT_b, (long)HHn * MMf * NNs, (long)MMf * NNs,
      ctxT, (long)HHn * DHd * MMf, (long)DHd * MMf, MMf,
      nullptr, 0, 0, nullptr, NNs, HHn);

  // attn[z,n,d] = dinv[z,n] * (qp[z,n,:] @ ctxT[z,d,:]^T), merged-head store
  dim3 gATT(DHd / 64, NNs / 128, ZZb);
  gemm_k<EPI_SCALEROW_BF16><<<gATT, 256, 0, stream>>>(
      qp_b, (long)HHn * NNs * MMf, (long)NNs * MMf, MMf,
      ctxT, (long)HHn * DHd * MMf, (long)DHd * MMf,
      attn_b, (long)NNs * DDm, DHd, DDm,
      dinv, (long)HHn * NNs, NNs, nullptr, MMf, HHn);

  // x1 = x + attn @ wo^T + bo  (fp32)
  gemm_k<EPI_BIAS_RES_F32><<<gQKV, 256, 0, stream>>>(
      attn_b, 0, 0, DDm, wo_b, 0, 0, x1, 0, 0, DDm, bo, 0, 0, x, DDm, 1);

  // LN2 -> h2
  ln_k<<<dim3(BNr / 8), 256, 0, stream>>>(x1, g2, bln2, h2_b);

  // ff1 = gelu(h2 @ w1^T + b1)  : [32768,512] x [2048,512]^T
  dim3 gFF1(FFd / 64, BNr / 128, 1);
  gemm_k<EPI_GELU_BF16><<<gFF1, 256, 0, stream>>>(
      h2_b, 0, 0, DDm, w1_b, 0, 0, ff1_b, 0, 0, FFd, pb1, 0, 0, nullptr, DDm, 1);

  // x2 = x1 + ff1 @ w2^T + b2  (bf16) : [32768,2048] x [512,2048]^T
  gemm_k<EPI_BIAS_RES_BF16><<<gQKV, 256, 0, stream>>>(
      ff1_b, 0, 0, FFd, w2_b, 0, 0, x2_b, 0, 0, DDm, pb2, 0, 0, x1, FFd, 1);

  // out = x2 @ wop^T + bop  (fp32 to d_out)
  gemm_k<EPI_BIAS_F32><<<gQKV, 256, 0, stream>>>(
      x2_b, 0, 0, DDm, wop_b, 0, 0, out, 0, 0, DDm, bop, 0, 0, nullptr, DDm, 1);

  (void)in_sizes; (void)n_in; (void)out_size; (void)ws_size;
}
